// MultiHeadLatentAttention_75445395522225
// MI455X (gfx1250) — compile-verified, matrix operand order checked by reading, not run on hardware
//
#include <hip/hip_runtime.h>
#include <hip/hip_bf16.h>
#include <math.h>

// Problem constants (from reference)
#define B_   2
#define T_   2048
#define C_   2048
#define H_   16
#define HD_  128
#define DKV_ 512
#define DQ_  1536
#define DR_  64
#define ROWS (B_ * T_)        // 4096 token rows
#define DHEAD (HD_ + DR_)     // 192 = per-head q/k dim
#define QK_LD (H_ * DHEAD)    // 3072 = row stride of assembled q/k

typedef __attribute__((ext_vector_type(16))) __bf16 v16bf;
typedef __attribute__((ext_vector_type(8)))  float  v8f;
typedef int v4i_t __attribute__((vector_size(16)));   // matches builtin proto

// ---------------------------------------------------------------- helpers ---
static __device__ __forceinline__ __bf16 f2bf(float f) {
    unsigned u = __float_as_uint(f);
    u += 0x7FFFu + ((u >> 16) & 1u);               // round-to-nearest-even
    unsigned short h = (unsigned short)(u >> 16);
    union { unsigned short s; __bf16 b; } cv; cv.s = h; return cv.b;
}

// 16-byte global -> LDS copy. Prefer the CDNA5 async data path (ASYNCcnt),
// fall back to a sync load + ds_store if the builtin is absent.
static __device__ __forceinline__ void async_copy16(const __bf16* g, __bf16* l) {
#if __has_builtin(__builtin_amdgcn_global_load_async_to_lds_b128)
    typedef __attribute__((address_space(1))) v4i_t* gvp;
    typedef __attribute__((address_space(3))) v4i_t* lvp;
    __builtin_amdgcn_global_load_async_to_lds_b128(
        (gvp)(void*)g, (lvp)(void*)l, 0, 0);
#else
    *(uint4*)l = *(const uint4*)g;
#endif
}

static __device__ __forceinline__ void async_wait() {
#if __has_builtin(__builtin_amdgcn_s_wait_asynccnt)
    __builtin_amdgcn_s_wait_asynccnt(0);
#elif __has_builtin(__builtin_amdgcn_global_load_async_to_lds_b128)
    asm volatile("s_wait_asynccnt 0" ::: "memory");
#endif
}

// Load a 16x32 bf16 WMMA A/B fragment from a row-major matrix (global or LDS).
// ISA layout (16-bit A 16x32): lanes 0-15 = row m, elems 0-7 -> K 0-7,
// elems 8-15 -> K 16-23; lanes 16-31 = row m, K 8-15 / 24-31.
// => per lane: two contiguous 16-byte loads at k0 + half*8 and +16.
static __device__ __forceinline__ v16bf load_frag(const __bf16* base, int ld,
                                                  int row0, int k0) {
    int lane = threadIdx.x & 31;
    int m    = lane & 15;
    int half = lane >> 4;
    const __bf16* p = base + (size_t)(row0 + m) * ld + k0 + half * 8;
    union { v16bf v; uint4 q[2]; } u;
    u.q[0] = *(const uint4*)(p);
    u.q[1] = *(const uint4*)(p + 16);
    return u.v;
}

static __device__ __forceinline__ v8f wmma_bf16(v16bf a, v16bf b, v8f c) {
    return __builtin_amdgcn_wmma_f32_16x16x32_bf16(
        /*neg_a=*/false, a, /*neg_b=*/false, b,
        /*c_mod=*/(short)0, c, /*reuse_a=*/false, /*reuse_b=*/false);
}

// ------------------------------------------------------ elementwise kernels ---
__global__ __launch_bounds__(256) void k_cvt(const float* __restrict__ x,
                                             __bf16* __restrict__ y, size_t n) {
    for (size_t i = (size_t)blockIdx.x * 256 + threadIdx.x; i < n;
         i += (size_t)gridDim.x * 256)
        y[i] = f2bf(x[i]);
}

// Wt[n][k] = W[k][n]  (f32 KxN row-major -> bf16 NxK row-major)
__global__ __launch_bounds__(256) void k_tconv(const float* __restrict__ W,
                                               __bf16* __restrict__ Wt,
                                               int K, int N) {
    size_t total = (size_t)K * N;
    for (size_t o = (size_t)blockIdx.x * 256 + threadIdx.x; o < total;
         o += (size_t)gridDim.x * 256) {
        size_t n = o / (size_t)K, k = o - n * (size_t)K;
        Wt[o] = f2bf(W[k * (size_t)N + n]);
    }
}

// ------------------------------------------------------------- GEMM (WMMA) ---
// C[M,N] (f32) = A[M,K] (bf16, row-major, lda) @ Bt[N,K]^T (bf16, row-major ldb)
// Block: 256 threads = 8 waves; wave w computes rows [(.y*8+w)*16, +16),
// cols [blockIdx.x*64, +64). The 64x32 B tile is shared by all 8 waves and is
// staged through double-buffered LDS with async global->LDS copies (one b128
// chunk per thread), overlapping the next tile's fetch with this tile's WMMAs.
__global__ __launch_bounds__(256) void k_gemm(const __bf16* __restrict__ A,
                                              const __bf16* __restrict__ Bt,
                                              float* __restrict__ C,
                                              int M, int N, int K,
                                              int lda, int ldb, int ldc) {
    __shared__ __bf16 bsh[2][64 * 32];           // 2 x 4 KB
    int wave = threadIdx.x >> 5;
    int row0 = (blockIdx.y * 8 + wave) * 16;
    int col0 = blockIdx.x * 64;
    // This thread's 16-byte chunk of the 64x32 B tile (256 threads * 16B = 4KB)
    int bo   = threadIdx.x * 16;                 // byte offset in tile
    int brow = bo >> 6;                          // 64 bytes (32 bf16) per row
    int bk   = (bo & 63) >> 1;                   // element offset in row

    auto stage = [&](int buf, int k0) {
        async_copy16(Bt + (size_t)(col0 + brow) * ldb + k0 + bk,
                     &bsh[buf][brow * 32 + bk]);
    };

    stage(0, 0);
    async_wait();
    __syncthreads();

    v8f acc[4] = {};
    for (int k0 = 0; k0 < K; k0 += 32) {
        int cur = (k0 >> 5) & 1;
        if (k0 + 32 < K) {
            stage(cur ^ 1, k0 + 32);             // prefetch next tile into LDS
            __builtin_prefetch(A + (size_t)row0 * lda + k0 + 32, 0, 1);
        }
        v16bf a = load_frag(A, lda, row0, k0);
        v16bf b0 = load_frag(&bsh[cur][0], 32,  0, 0);
        v16bf b1 = load_frag(&bsh[cur][0], 32, 16, 0);
        v16bf b2 = load_frag(&bsh[cur][0], 32, 32, 0);
        v16bf b3 = load_frag(&bsh[cur][0], 32, 48, 0);
        acc[0] = wmma_bf16(a, b0, acc[0]);
        acc[1] = wmma_bf16(a, b1, acc[1]);
        acc[2] = wmma_bf16(a, b2, acc[2]);
        acc[3] = wmma_bf16(a, b3, acc[3]);
        async_wait();                            // own async chunk landed
        __syncthreads();                         // whole tile landed, reads done
    }
    int lane = threadIdx.x & 31;
    int n = lane & 15, rb = (lane >> 4) * 8;     // C layout: lane=col, vgpr=row
#pragma unroll
    for (int j = 0; j < 4; ++j)
#pragma unroll
        for (int r = 0; r < 8; ++r)
            C[(size_t)(row0 + rb + r) * ldc + col0 + j * 16 + n] = acc[j][r];
}

// ---------------------------------------------------------------- LayerNorm ---
__global__ __launch_bounds__(256) void k_ln(const float* __restrict__ X,
                                            const float* __restrict__ g,
                                            const float* __restrict__ bta,
                                            __bf16* __restrict__ Y, int N) {
    __shared__ float red[256];
    int row = blockIdx.x;
    const float* x = X + (size_t)row * N;
    float s = 0.f, ss = 0.f;
    for (int i = threadIdx.x; i < N; i += 256) { float v = x[i]; s += v; ss += v * v; }
    red[threadIdx.x] = s;  __syncthreads();
    for (int st = 128; st > 0; st >>= 1) {
        if ((int)threadIdx.x < st) red[threadIdx.x] += red[threadIdx.x + st];
        __syncthreads();
    }
    float mean = red[0] / N;  __syncthreads();
    red[threadIdx.x] = ss; __syncthreads();
    for (int st = 128; st > 0; st >>= 1) {
        if ((int)threadIdx.x < st) red[threadIdx.x] += red[threadIdx.x + st];
        __syncthreads();
    }
    float var = red[0] / N - mean * mean;
    float inv = rsqrtf(var + 1e-5f);
    for (int i = threadIdx.x; i < N; i += 256)
        Y[(size_t)row * N + i] = f2bf((x[i] - mean) * inv * g[i] + bta[i]);
}

// ---------------------------------------------- assemble q / k (with rope) ---
__global__ __launch_bounds__(256) void k_assemble(const float* __restrict__ cmain,
                                                  const float* __restrict__ rp,
                                                  const float* __restrict__ cosb,
                                                  const float* __restrict__ sinb,
                                                  __bf16* __restrict__ out,
                                                  int h_stride, int rope_ld,
                                                  float scale) {
    int row = blockIdx.x;
    int t   = row & (T_ - 1);
    for (int idx = threadIdx.x; idx < H_ * DHEAD; idx += 256) {
        int h = idx / DHEAD, d = idx - h * DHEAD;
        float v;
        if (d < HD_) {
            v = cmain[(size_t)row * C_ + h * HD_ + d];
        } else {
            int dr = d - HD_;
            const float* rrow = rp + (size_t)row * rope_ld + h * h_stride;
            float c  = cosb[t * DR_ + dr];
            float s  = sinb[t * DR_ + dr];
            float x1 = rrow[dr];
            float x2 = (dr < DR_ / 2) ? -rrow[dr + DR_ / 2] : rrow[dr - DR_ / 2];
            v = x1 * c + x2 * s;                 // x*cos + rotate_half(x)*sin
        }
        out[(size_t)row * QK_LD + idx] = f2bf(v * scale);
    }
}

// vt[b][h][d][t] = vc[(b*T+t)][h*128+d]   (bf16, so P@V B-fragments are rows)
__global__ __launch_bounds__(256) void k_vtrans(const float* __restrict__ vc,
                                                __bf16* __restrict__ vt) {
    size_t total = (size_t)B_ * H_ * HD_ * T_;
    for (size_t o = (size_t)blockIdx.x * 256 + threadIdx.x; o < total;
         o += (size_t)gridDim.x * 256) {
        int t  = (int)(o & (T_ - 1));
        size_t rest = o >> 11;                   // / T_
        int d  = (int)(rest & (HD_ - 1));
        int bh = (int)(rest >> 7);               // / HD_
        int b  = bh >> 4, h = bh & (H_ - 1);
        vt[o] = f2bf(vc[(size_t)(b * T_ + t) * C_ + h * HD_ + d]);
    }
}

// --------------------------------------------------- flash attention (WMMA) ---
// Block = 8 waves = 8 consecutive 16-row Q tiles of ONE (b,h); K and V tiles
// for the current 32 keys are shared by all waves, staged via async copies
// into double-buffered LDS. Online softmax entirely in WMMA C-layout; P is
// repacked C-layout -> A-layout through a per-wave LDS tile.
__global__ __launch_bounds__(256) void k_attn(const __bf16* __restrict__ qb,
                                              const __bf16* __restrict__ kb,
                                              const __bf16* __restrict__ vt,
                                              __bf16* __restrict__ ob) {
    __shared__ __bf16 ksh[2][32 * DHEAD];        // 2 x 12 KB (32 keys x 192)
    __shared__ __bf16 vsh[2][HD_ * 32];          // 2 x  8 KB (128 d x 32 keys)
    __shared__ __bf16 psh[8][16 * 32];           //      8 KB per-wave P tiles
    int wave = threadIdx.x >> 5, lane = threadIdx.x & 31;
    int bh = blockIdx.x >> 4;                    // 16 blocks per (b,h)
    int b  = bh >> 4, h = bh & (H_ - 1);
    int qt = (blockIdx.x & 15) * 8 + wave;       // this wave's q tile
    int qrow0 = b * T_ + qt * 16;

    const __bf16* qbase = qb + (size_t)h * DHEAD;
    const __bf16* kgb   = kb + (size_t)h * DHEAD;            // row stride QK_LD
    const __bf16* vgb   = vt + (size_t)(b * H_ + h) * HD_ * T_;  // [128][T]

    // Stage K tile: 32 rows x 192 elems = 768 16B chunks (3 per thread).
    auto stageK = [&](int buf, int t0) {
        for (int c = threadIdx.x; c < 768; c += 256) {
            int r = c / 24, e = (c % 24) * 8;    // 24 chunks per 192-elem row
            async_copy16(kgb + (size_t)(b * T_ + t0 + r) * QK_LD + e,
                         &ksh[buf][r * DHEAD + e]);
        }
    };
    // Stage V tile: 128 rows x 32 elems = 512 16B chunks (2 per thread).
    auto stageV = [&](int buf, int t0) {
        for (int c = threadIdx.x; c < 512; c += 256) {
            int r = c / 4, e = (c % 4) * 8;      // 4 chunks per 32-elem row
            async_copy16(vgb + (size_t)r * T_ + t0 + e,
                         &vsh[buf][r * 32 + e]);
        }
    };

    v16bf qf[6];
#pragma unroll
    for (int j = 0; j < 6; ++j) qf[j] = load_frag(qbase, QK_LD, qrow0, j * 32);

    float mrow[8], lrow[8];
#pragma unroll
    for (int r = 0; r < 8; ++r) { mrow[r] = -3.0e38f; lrow[r] = 0.f; }
    v8f o[8] = {};
    int n = lane & 15, rb = (lane >> 4) * 8;
    __bf16* pb = &psh[wave][0];

    stageK(0, 0); stageV(0, 0);
    async_wait();
    __syncthreads();

    for (int t0 = 0; t0 < T_; t0 += 32) {
        int cur = (t0 >> 5) & 1;
        if (t0 + 32 < T_) { stageK(cur ^ 1, t0 + 32); stageV(cur ^ 1, t0 + 32); }

        // S = Q @ K^T for 32 keys: preload all 12 K fragments, then 12 WMMAs.
        v16bf kf[12];
#pragma unroll
        for (int j = 0; j < 6; ++j) {
            kf[2 * j]     = load_frag(&ksh[cur][0], DHEAD, 0,  j * 32);
            kf[2 * j + 1] = load_frag(&ksh[cur][0], DHEAD, 16, j * 32);
        }
        v8f s0 = {}, s1 = {};
#pragma unroll
        for (int j = 0; j < 6; ++j) {
            s0 = wmma_bf16(qf[j], kf[2 * j],     s0);
            s1 = wmma_bf16(qf[j], kf[2 * j + 1], s1);
        }

        // Online softmax in C-layout (lane = col of rows rb..rb+7); row
        // reductions are shfl_xor trees over each 16-lane half.
#pragma unroll
        for (int r = 0; r < 8; ++r) {
            float v = fmaxf(s0[r], s1[r]);
            v = fmaxf(v, __shfl_xor(v, 8, 32));
            v = fmaxf(v, __shfl_xor(v, 4, 32));
            v = fmaxf(v, __shfl_xor(v, 2, 32));
            v = fmaxf(v, __shfl_xor(v, 1, 32));
            float mn = fmaxf(mrow[r], v);
            float p0 = __expf(s0[r] - mn);
            float p1 = __expf(s1[r] - mn);
            float rs = p0 + p1;
            rs += __shfl_xor(rs, 8, 32);
            rs += __shfl_xor(rs, 4, 32);
            rs += __shfl_xor(rs, 2, 32);
            rs += __shfl_xor(rs, 1, 32);
            float corr = __expf(mrow[r] - mn);
            lrow[r] = lrow[r] * corr + rs;
            mrow[r] = mn;
#pragma unroll
            for (int j = 0; j < 8; ++j) o[j][r] *= corr;
            pb[(rb + r) * 32 + n]      = f2bf(p0);   // C-layout -> row-major LDS
            pb[(rb + r) * 32 + 16 + n] = f2bf(p1);
        }
        asm volatile("s_wait_dscnt 0" ::: "memory"); // P stores visible to wave

        v16bf pf = load_frag(pb, 32, 0, 0);          // repacked A fragment
        v16bf vf[8];
#pragma unroll
        for (int j = 0; j < 8; ++j)
            vf[j] = load_frag(&vsh[cur][0], 32, j * 16, 0);
#pragma unroll
        for (int j = 0; j < 8; ++j) o[j] = wmma_bf16(pf, vf[j], o[j]);

        async_wait();                                // next K/V tile landed
        __syncthreads();                             // all waves done with cur
    }
#pragma unroll
    for (int j = 0; j < 8; ++j)
#pragma unroll
        for (int r = 0; r < 8; ++r)
            ob[(size_t)(qrow0 + rb + r) * C_ + h * HD_ + j * 16 + n] =
                f2bf(o[j][r] / lrow[r]);
}

// -------------------------------------------------------------------- host ---
extern "C" void kernel_launch(void* const* d_in, const int* in_sizes, int n_in,
                              void* d_out, int out_size, void* d_ws, size_t ws_size,
                              hipStream_t stream) {
    (void)in_sizes; (void)n_in; (void)out_size; (void)ws_size;
    const float* x       = (const float*)d_in[0];
    const float* cosb    = (const float*)d_in[1];
    const float* sinb    = (const float*)d_in[2];
    const float* Wdkv    = (const float*)d_in[3];
    const float* kvg     = (const float*)d_in[4];
    const float* kvb     = (const float*)d_in[5];
    const float* Wupk    = (const float*)d_in[6];
    const float* Wupv    = (const float*)d_in[7];
    const float* Wdq     = (const float*)d_in[8];
    const float* qg      = (const float*)d_in[9];
    const float* qbeta   = (const float*)d_in[10];
    const float* Wupq    = (const float*)d_in[11];
    const float* Wrq     = (const float*)d_in[12];
    const float* Wrk     = (const float*)d_in[13];
    const float* Wout    = (const float*)d_in[14];
    float* out = (float*)d_out;

    // Bump allocator over d_ws (total ~288 MiB).
    char* w = (char*)d_ws;
    auto alloc = [&](size_t bytes) -> void* {
        void* p = (void*)w; w += (bytes + 255) & ~(size_t)255; return p;
    };
    __bf16* xb     = (__bf16*)alloc((size_t)ROWS * C_ * 2);
    __bf16* wdkv_t = (__bf16*)alloc((size_t)DKV_ * C_ * 2);
    __bf16* wdq_t  = (__bf16*)alloc((size_t)DQ_ * C_ * 2);
    __bf16* wupk_t = (__bf16*)alloc((size_t)C_ * DKV_ * 2);
    __bf16* wupv_t = (__bf16*)alloc((size_t)C_ * DKV_ * 2);
    __bf16* wupq_t = (__bf16*)alloc((size_t)C_ * DQ_ * 2);
    __bf16* wrq_t  = (__bf16*)alloc((size_t)(H_ * DR_) * C_ * 2);
    __bf16* wrk_t  = (__bf16*)alloc((size_t)DR_ * C_ * 2);
    __bf16* wout_t = (__bf16*)alloc((size_t)C_ * C_ * 2);
    float*  ckv_f  = (float*) alloc((size_t)ROWS * DKV_ * 4);
    float*  cq_f   = (float*) alloc((size_t)ROWS * DQ_ * 4);
    __bf16* ckv_b  = (__bf16*)alloc((size_t)ROWS * DKV_ * 2);
    __bf16* cq_b   = (__bf16*)alloc((size_t)ROWS * DQ_ * 2);
    float*  qc_f   = (float*) alloc((size_t)ROWS * C_ * 4);
    float*  kc_f   = (float*) alloc((size_t)ROWS * C_ * 4);
    float*  vc_f   = (float*) alloc((size_t)ROWS * C_ * 4);
    float*  qr_f   = (float*) alloc((size_t)ROWS * H_ * DR_ * 4);
    float*  kr_f   = (float*) alloc((size_t)ROWS * DR_ * 4);
    __bf16* qbb    = (__bf16*)alloc((size_t)ROWS * QK_LD * 2);
    __bf16* kbb    = (__bf16*)alloc((size_t)ROWS * QK_LD * 2);
    __bf16* vtb    = (__bf16*)alloc((size_t)B_ * H_ * HD_ * T_ * 2);
    __bf16* aob    = (__bf16*)alloc((size_t)ROWS * C_ * 2);

    // 1) convert activations + transpose-convert weights to bf16
    k_cvt<<<2048, 256, 0, stream>>>(x, xb, (size_t)ROWS * C_);
    k_tconv<<<2048, 256, 0, stream>>>(Wdkv, wdkv_t, C_, DKV_);
    k_tconv<<<2048, 256, 0, stream>>>(Wdq,  wdq_t,  C_, DQ_);
    k_tconv<<<2048, 256, 0, stream>>>(Wupk, wupk_t, DKV_, C_);
    k_tconv<<<2048, 256, 0, stream>>>(Wupv, wupv_t, DKV_, C_);
    k_tconv<<<2048, 256, 0, stream>>>(Wupq, wupq_t, DQ_, C_);
    k_tconv<<<2048, 256, 0, stream>>>(Wrq,  wrq_t,  C_, H_ * DR_);
    k_tconv<<<512,  256, 0, stream>>>(Wrk,  wrk_t,  C_, DR_);
    k_tconv<<<2048, 256, 0, stream>>>(Wout, wout_t, C_, C_);

    // 2) down-projections + LN
    k_gemm<<<dim3(DKV_ / 64, ROWS / 128), 256, 0, stream>>>(xb, wdkv_t, ckv_f,
        ROWS, DKV_, C_, C_, C_, DKV_);
    k_gemm<<<dim3(DQ_ / 64, ROWS / 128), 256, 0, stream>>>(xb, wdq_t, cq_f,
        ROWS, DQ_, C_, C_, C_, DQ_);
    k_ln<<<ROWS, 256, 0, stream>>>(ckv_f, kvg, kvb, ckv_b, DKV_);
    k_ln<<<ROWS, 256, 0, stream>>>(cq_f,  qg,  qbeta, cq_b, DQ_);

    // 3) up-projections + rope projections
    k_gemm<<<dim3(C_ / 64, ROWS / 128), 256, 0, stream>>>(cq_b,  wupq_t, qc_f,
        ROWS, C_, DQ_, DQ_, DQ_, C_);
    k_gemm<<<dim3(C_ / 64, ROWS / 128), 256, 0, stream>>>(ckv_b, wupk_t, kc_f,
        ROWS, C_, DKV_, DKV_, DKV_, C_);
    k_gemm<<<dim3(C_ / 64, ROWS / 128), 256, 0, stream>>>(ckv_b, wupv_t, vc_f,
        ROWS, C_, DKV_, DKV_, DKV_, C_);
    k_gemm<<<dim3((H_ * DR_) / 64, ROWS / 128), 256, 0, stream>>>(xb, wrq_t, qr_f,
        ROWS, H_ * DR_, C_, C_, C_, H_ * DR_);
    k_gemm<<<dim3(DR_ / 64, ROWS / 128), 256, 0, stream>>>(xb, wrk_t, kr_f,
        ROWS, DR_, C_, C_, C_, DR_);

    // 4) assemble q/k (rope + concat; 1/sqrt(192) folded into q), transpose v
    const float qscale = 0.07216878364870323f;   // 1/sqrt(HD+DR)
    k_assemble<<<ROWS, 256, 0, stream>>>(qc_f, qr_f, cosb, sinb, qbb,
                                         DR_, H_ * DR_, qscale);
    k_assemble<<<ROWS, 256, 0, stream>>>(kc_f, kr_f, cosb, sinb, kbb,
                                         0, DR_, 1.0f);
    k_vtrans<<<2048, 256, 0, stream>>>(vc_f, vtb);

    // 5) attention: 16 blocks per (b,h); 8 q-tiles per block
    k_attn<<<(B_ * H_ * (T_ / 16)) / 8, 256, 0, stream>>>(qbb, kbb, vtb, aob);

    // 6) output projection straight into d_out (f32)
    k_gemm<<<dim3(C_ / 64, ROWS / 128), 256, 0, stream>>>(aob, wout_t, out,
        ROWS, C_, C_, C_, C_, C_);
}
